// SparseAttention_78864189489863
// MI455X (gfx1250) — matrix operand order repaired and checked by reading, not compiled, not run on hardware
//
#include <hip/hip_runtime.h>

// Sparse attention (uniform CSR, R=128 nz/row) for MI455X / gfx1250.
// One wave32 per (batch,row): gather K/V rows as bf16 into LDS via
// GLOBAL_LOAD_ASYNC_TO_LDS_B128 (ASYNCcnt), SDDMM and SPMM via
// v_wmma_f32_16x16x32_bf16, wave-cooperative softmax in f32 overlapped
// with the async V gather.

#define BATCH 8
#define SEQ   4096
#define HD    64
#define RNZ   128
#define NWAVE 2   // waves (rows) per workgroup; LDS = NWAVE * ~17.3KB

typedef __attribute__((ext_vector_type(16))) __bf16 v16bf;
typedef __attribute__((ext_vector_type(8)))  float  v8f;

union Frag {
  unsigned int   u[8];
  unsigned short h[16];
  v16bf          v;
};

__device__ __forceinline__ unsigned short f2bf(float f) {
  unsigned int u = __float_as_uint(f);
  u = (u + 0x7FFFu + ((u >> 16) & 1u)) >> 16;   // round-to-nearest-even
  return (unsigned short)u;
}
__device__ __forceinline__ unsigned int pack2(float lo, float hi) {
  return (unsigned int)f2bf(lo) | ((unsigned int)f2bf(hi) << 16);
}

// Flat LDS addresses: aperture tag in [63:32], byte offset in [31:0].
__device__ __forceinline__ unsigned lds_off(const void* p) {
  return (unsigned)(unsigned long long)p;
}
// Async copy 16B global -> LDS (GV mode, no VGPR staging, ASYNCcnt).
__device__ __forceinline__ void async_ld_b128(unsigned lds_byte, const void* g) {
  asm volatile("global_load_async_to_lds_b128 %0, %1, off"
               :: "v"(lds_byte), "v"(g) : "memory");
}
__device__ __forceinline__ void wait_async0() {
  asm volatile("s_wait_asynccnt 0" ::: "memory");
}
__device__ __forceinline__ void wait_ds0() {
  asm volatile("s_wait_dscnt 0" ::: "memory");
}

// ---------------- f32 -> bf16 pre-conversion (fills d_ws) ----------------
__global__ void f32_to_bf16_kernel(const float* __restrict__ src,
                                   unsigned short* __restrict__ dst, int n) {
  int i = (blockIdx.x * blockDim.x + threadIdx.x) * 4;
  const int stride = gridDim.x * blockDim.x * 4;
  for (; i < n; i += stride) {
    float4 f = *(const float4*)(src + i);
    unsigned int* d = (unsigned int*)(dst + i);
    d[0] = pack2(f.x, f.y);
    d[1] = pack2(f.z, f.w);
  }
}

// ---------------- main kernel: one wave per (b, row) ----------------
__global__ __launch_bounds__(NWAVE * 32)
void sparse_attn_kernel(const int* __restrict__ row_offsets,
                        const int* __restrict__ colidx,
                        const float* __restrict__ q3d,
                        const float* __restrict__ k3d,
                        const float* __restrict__ v3d,
                        const unsigned short* __restrict__ qbf,
                        const unsigned short* __restrict__ kbf,
                        const unsigned short* __restrict__ vbf,
                        float* __restrict__ out,
                        int use_bf16) {
  __shared__ unsigned short kv[NWAVE][RNZ * HD];  // gathered K rows, then V rows (bf16)
  __shared__ unsigned short qs[NWAVE][HD];        // q row (bf16)
  __shared__ float          lw[NWAVE][RNZ];       // f32 logits
  __shared__ unsigned short wb[NWAVE][RNZ];       // softmax weights (bf16)

  const int lane = threadIdx.x & 31;
  const int wv   = threadIdx.x >> 5;
  const int g    = blockIdx.x * NWAVE + wv;       // global row id in [0, B*SEQ)
  const int b    = g / SEQ;
  const int row  = g % SEQ;
  const int half = lane >> 4;                     // 0: lanes 0-15, 1: lanes 16-31
  const int mrow = lane & 15;

  const int start = row_offsets[row];             // uniform CSR: start = row*RNZ

  unsigned short* KV = kv[wv];
  unsigned short* QS = qs[wv];
  float*          LW = lw[wv];
  unsigned short* WB = wb[wv];

  // ---- stage Q row (64 values) into LDS as bf16: 2 values / lane ----
  {
    const size_t qoff = ((size_t)b * SEQ + row) * HD + (size_t)lane * 2;
    if (use_bf16) {
      *(unsigned int*)(QS + lane * 2) = *(const unsigned int*)(qbf + qoff);
    } else {
      *(unsigned int*)(QS + lane * 2) = pack2(q3d[qoff], q3d[qoff + 1]);
    }
  }

  // ---- gather 128 K rows into LDS (bf16, row-major 128x64) ----
  if (use_bf16) {
    // Async path: 4 rows / iter, each lane moves 16B directly to LDS.
    for (int it = 0; it < 32; ++it) {
      const int r   = it * 4 + (lane >> 3);       // nonzero slot 0..127
      const int c   = lane & 7;                   // 8-elem chunk 0..7
      const int col = colidx[start + r];
      async_ld_b128(lds_off(KV + r * HD + c * 8),
                    kbf + ((size_t)b * SEQ + col) * HD + (size_t)c * 8);
    }
    wait_async0();                                // LDS tile complete
  } else {
    for (int it = 0; it < 64; ++it) {
      const int r   = it * 2 + half;
      const int c   = mrow;
      const int col = colidx[start + r];
      const size_t src = ((size_t)b * SEQ + col) * HD + (size_t)c * 4;
      float4 f = *(const float4*)(k3d + src);
      unsigned int* d = (unsigned int*)(KV + r * HD + c * 4);
      d[0] = pack2(f.x, f.y); d[1] = pack2(f.z, f.w);
    }
  }

  // ---- B-fragments for SDDMM: B[k][n] = q[k] (broadcast over n) ----
  // B k order chosen here (matches A): element i -> K = 16*half + i (pairs contiguous).
  Frag qf[2];
#pragma unroll
  for (int kc = 0; kc < 2; ++kc)
#pragma unroll
    for (int p = 0; p < 8; ++p)
      qf[kc].u[p] = *(const unsigned int*)(QS + kc * 32 + 16 * half + 2 * p);

  // ---- SDDMM: logits[j] = <Q[row], K[col j]>, 8 M-tiles x 2 K-chunks ----
  // A 16x32 operand, k order chosen here (matches B): lane M = lane%16; element i ->
  //   K = 16*half + i; pairs are contiguous.
#pragma unroll
  for (int t = 0; t < 8; ++t) {
    v8f acc = {};
#pragma unroll
    for (int kc = 0; kc < 2; ++kc) {
      Frag af;
      const unsigned short* base = KV + (t * 16 + mrow) * HD + kc * 32;
#pragma unroll
      for (int p = 0; p < 8; ++p) {
        const int kl = 16 * half + 2 * p;
        af.u[p] = *(const unsigned int*)(base + kl);
      }
      acc = __builtin_amdgcn_wmma_f32_16x16x32_bf16(false, af.v, false, qf[kc].v,
                                                    (short)0, acc, false, false);
    }
    // D layout: VGPR r holds M = r + 8*half, replicated over N -> lanes 0/16
    // spill the 16 logits of this tile.
    if (mrow == 0) {
#pragma unroll
      for (int r = 0; r < 8; ++r)
        LW[t * 16 + 8 * half + r] = acc[r];
    }
  }

  // ---- start async V gather (overwrites KV) before softmax ----
  if (use_bf16) {
    wait_ds0();   // all SDDMM fragment reads of KV retired before overwrite
    for (int it = 0; it < 32; ++it) {
      const int r   = it * 4 + (lane >> 3);
      const int c   = lane & 7;
      const int col = colidx[start + r];
      async_ld_b128(lds_off(KV + r * HD + c * 8),
                    vbf + ((size_t)b * SEQ + col) * HD + (size_t)c * 8);
    }
    // V copy now in flight; softmax below only touches LW/WB.
  }

  // ---- softmax over 128 logits (wave-cooperative, f32) ----
  {
    float vv[4];
#pragma unroll
    for (int e = 0; e < 4; ++e) vv[e] = LW[lane * 4 + e];
    float mx = fmaxf(fmaxf(vv[0], vv[1]), fmaxf(vv[2], vv[3]));
#pragma unroll
    for (int off = 16; off > 0; off >>= 1)
      mx = fmaxf(mx, __shfl_xor(mx, off, 32));
    float s = 0.f;
#pragma unroll
    for (int e = 0; e < 4; ++e) { vv[e] = __expf(vv[e] - mx); s += vv[e]; }
#pragma unroll
    for (int off = 16; off > 0; off >>= 1)
      s += __shfl_xor(s, off, 32);
    const float inv = 1.0f / s;
#pragma unroll
    for (int e = 0; e < 4; ++e) WB[lane * 4 + e] = f2bf(vv[e] * inv);
  }

  // ---- finish V gather ----
  if (use_bf16) {
    wait_async0();
  } else {
    for (int it = 0; it < 64; ++it) {
      const int r   = it * 2 + half;
      const int c   = mrow;
      const int col = colidx[start + r];
      const size_t src = ((size_t)b * SEQ + col) * HD + (size_t)c * 4;
      float4 f = *(const float4*)(v3d + src);
      unsigned int* d = (unsigned int*)(KV + r * HD + c * 4);
      d[0] = pack2(f.x, f.y); d[1] = pack2(f.z, f.w);
    }
  }

  // ---- A-fragments for SPMM: A[m][k] = w[k] (broadcast over m) ----
  Frag wf[4];
#pragma unroll
  for (int kc = 0; kc < 4; ++kc)
#pragma unroll
    for (int p = 0; p < 8; ++p) {
      const int kl = 16 * half + 2 * p;
      wf[kc].u[p] = *(const unsigned int*)(WB + kc * 32 + kl);
    }

  // ---- SPMM: out[n] = sum_j w[j] * V[col j][n], 4 N-tiles x 4 K-chunks ----
#pragma unroll
  for (int nt = 0; nt < 4; ++nt) {
    v8f acc = {};
#pragma unroll
    for (int kc = 0; kc < 4; ++kc) {
      Frag bfr;  // B element i -> K = kc*32 + 16*half + i, N = mrow
#pragma unroll
      for (int i = 0; i < 16; ++i)
        bfr.h[i] = KV[(kc * 32 + 16 * half + i) * HD + nt * 16 + mrow];
      acc = __builtin_amdgcn_wmma_f32_16x16x32_bf16(false, wf[kc].v, false, bfr.v,
                                                    (short)0, acc, false, false);
    }
    // All M rows of D are identical; lane L (half==0) holds out[nt*16 + L].
    if (half == 0)
      out[((size_t)b * SEQ + row) * HD + nt * 16 + mrow] = acc[0];
  }
}

extern "C" void kernel_launch(void* const* d_in, const int* in_sizes, int n_in,
                              void* d_out, int out_size, void* d_ws, size_t ws_size,
                              hipStream_t stream) {
  (void)in_sizes; (void)n_in; (void)out_size;
  // setup_inputs order: row_indices, row_offsets, column_indices, q3d, k3d, v3d, values
  const int*   row_offsets = (const int*)d_in[1];
  const int*   column_idx  = (const int*)d_in[2];
  const float* q3d         = (const float*)d_in[3];
  const float* k3d         = (const float*)d_in[4];
  const float* v3d         = (const float*)d_in[5];
  float*       out         = (float*)d_out;

  const size_t nElem   = (size_t)BATCH * SEQ * HD;         // 2,097,152 per tensor
  const size_t bfBytes = nElem * sizeof(unsigned short);   // 4 MB per tensor
  const int use_bf16   = (ws_size >= 3 * bfBytes) ? 1 : 0;

  unsigned short* qbf = (unsigned short*)d_ws;
  unsigned short* kbf = qbf + nElem;
  unsigned short* vbf = kbf + nElem;

  if (use_bf16) {
    dim3 cg(1024), cb(256);
    f32_to_bf16_kernel<<<cg, cb, 0, stream>>>(q3d, qbf, (int)nElem);
    f32_to_bf16_kernel<<<cg, cb, 0, stream>>>(k3d, kbf, (int)nElem);
    f32_to_bf16_kernel<<<cg, cb, 0, stream>>>(v3d, vbf, (int)nElem);
  }

  const int rows   = BATCH * SEQ;          // 32768
  const int blocks = rows / NWAVE;         // 16384, exact
  sparse_attn_kernel<<<blocks, NWAVE * 32, 0, stream>>>(
      row_offsets, column_idx, q3d, k3d, v3d, qbf, kbf, vbf, out, use_bf16);
}
